// NTM_4922032521896
// MI455X (gfx1250) — compile-verified, matrix operand order checked by reading, not run on hardware
//
#include <hip/hip_runtime.h>
#include <stdint.h>

#define B_   64
#define T_   64
#define NI_  2048
#define NH_  2048
#define MB_  4
#define MN_  128
#define MW_  20
#define MWP_ 32   // MW padded to one WMMA K-step (zeros beyond 20)

typedef __attribute__((ext_vector_type(16))) __bf16   v16bf;
typedef __attribute__((ext_vector_type(8)))  float    v8f;
typedef __attribute__((ext_vector_type(4)))  uint32_t u32x4;

// Load a 16x32 bf16 WMMA A/B fragment for this lane.
// Caller passes p = base + (tile_row + (lane&15))*ld + (lane>>4)*8.
// Low 8 elements = K {0..7} (or {8..15} for upper lanes), high 8 = +16.
__device__ inline v16bf ldfrag(const __bf16* p) {
  union { u32x4 u[2]; v16bf v; } x;
  x.u[0] = *(const u32x4*)(p);
  x.u[1] = *(const u32x4*)(p + 16);
  return x.v;
}

__device__ inline v8f wmma_bf16(v16bf a, v16bf b, v8f c) {
  return __builtin_amdgcn_wmma_f32_16x16x32_bf16(false, a, false, b, (short)0, c, false, false);
}

// ---------------- phase 0: converts / transposes ----------------

// x [B,T,NI] f32 -> Xbf [T*B, NI] bf16 (row = t*64 + b)
__global__ void k_cvt_x(const float* __restrict__ x, __bf16* __restrict__ xb) {
  int row = blockIdx.x;
  int t = row >> 6, b = row & 63;
  const float* src = x + ((size_t)b * T_ + t) * NI_;
  __bf16* dst = xb + (size_t)row * NI_;
  for (int i = threadIdx.x; i < NI_; i += blockDim.x) dst[i] = (__bf16)src[i];
}

__global__ void k_cvt(const float* __restrict__ s, __bf16* __restrict__ d, int n) {
  int i = blockIdx.x * blockDim.x + threadIdx.x;
  if (i < n) d[i] = (__bf16)s[i];
}

// dst[c*Rd + r] = (r<R && c<C) ? src[r*C + c] : 0   (tiled transpose + cvt, pads with zero)
__global__ void k_transpose_cvt(const float* __restrict__ src, __bf16* __restrict__ dst,
                                int R, int C, int Rd, int Cd) {
  __shared__ float tile[32][33];
  int c0 = blockIdx.x * 32, r0 = blockIdx.y * 32;
  int tx = threadIdx.x, ty = threadIdx.y;
  for (int i = 0; i < 32; i += 8) {
    int r = r0 + ty + i, c = c0 + tx;
    tile[ty + i][tx] = (r < R && c < C) ? src[(size_t)r * C + c] : 0.0f;
  }
  __syncthreads();
  for (int i = 0; i < 32; i += 8) {
    int c = c0 + ty + i, r = r0 + tx;
    if (c < Cd && r < Rd) dst[(size_t)c * Rd + r] = (__bf16)tile[tx][ty + i];
  }
}

__global__ void k_bank_norms(const float* __restrict__ mem, float* __restrict__ norms) {
  int i = blockIdx.x * blockDim.x + threadIdx.x;
  if (i < MB_ * MN_) {
    const float* p = mem + (size_t)i * MW_;
    float s = 0.f;
    for (int c = 0; c < MW_; ++c) s += p[c] * p[c];
    norms[i] = sqrtf(s);
  }
}

// ---------------- big GEMM: C[M,N] = A[M,K] @ BT[N,K]^T + bias ----------------
// mode 0: store f32 (Z = x@W_in + b_in)
// mode 1: clip(v,0,1), remap row m=t*64+b -> orow=b*64+t (out = [B,T,NI])
__global__ void __launch_bounds__(256) k_gemm(
    const __bf16* __restrict__ A, const __bf16* __restrict__ BT,
    const float* __restrict__ bias, float* __restrict__ out,
    int M, int N, int K, int mode) {
  int lane = threadIdx.x & 31;
  int wid  = blockIdx.x * (blockDim.x >> 5) + (threadIdx.x >> 5);
  int tiles_m = M >> 6;
  int tm = (wid % tiles_m) << 6;
  int tn = (wid / tiles_m) << 6;
  int lm = lane & 15, lh = lane >> 4;

  const __bf16* Ab = A  + (size_t)(tm + lm) * K + lh * 8;
  const __bf16* Bb = BT + (size_t)(tn + lm) * K + lh * 8;

  v8f acc[4][4] = {};
  for (int k = 0; k < K; k += 32) {
    __builtin_prefetch((const void*)(Ab + k + 512), 0, 0);
    __builtin_prefetch((const void*)(Bb + k + 512), 0, 0);
    v16bf a[4], b[4];
#pragma unroll
    for (int i = 0; i < 4; ++i) a[i] = ldfrag(Ab + (size_t)(i * 16) * K + k);
#pragma unroll
    for (int j = 0; j < 4; ++j) b[j] = ldfrag(Bb + (size_t)(j * 16) * K + k);
#pragma unroll
    for (int i = 0; i < 4; ++i)
#pragma unroll
      for (int j = 0; j < 4; ++j)
        acc[i][j] = wmma_bf16(a[i], b[j], acc[i][j]);
  }

#pragma unroll
  for (int i = 0; i < 4; ++i)
#pragma unroll
    for (int j = 0; j < 4; ++j) {
      int col = tn + j * 16 + lm;
      float bv = bias[col];
#pragma unroll
      for (int r = 0; r < 8; ++r) {
        int m = tm + i * 16 + r + (lh << 3);   // C layout: VGPR r -> M=r / r+8
        float v = acc[i][j][r] + bv;
        if (mode == 0) {
          out[(size_t)m * N + col] = v;
        } else {
          v = fminf(fmaxf(v, 0.0f), 1.0f);
          int orow = ((m & 63) << 6) + (m >> 6);
          out[(size_t)orow * N + col] = v;
        }
      }
    }
}

// ---------------- phase 2: sequential scan (one persistent workgroup) ----------------
__global__ void __launch_bounds__(1024) k_recurrence(
    const __bf16* __restrict__ H0, __bf16* __restrict__ H,
    const __bf16* __restrict__ WkeyT,   // [MWP_, NH]
    const __bf16* __restrict__ WreadT,  // [NH, MWP_]
    const float* __restrict__ bkey, const float* __restrict__ bread,
    const float* __restrict__ Z,        // [T*B, NH] f32
    const float* __restrict__ mem,      // [4,128,20]
    const float* __restrict__ norms) {  // [4*128]
  __shared__ float  kbuf[B_][MWP_];        // 8 KB
  __shared__ float  partial[8][4][256];    // 32 KB (k-split partials)
  __shared__ __bf16 rbuf[B_][MWP_];        // 4 KB

  int tid = threadIdx.x;
  int wave = tid >> 5, lane = tid & 31;
  int lm = lane & 15, lh = lane >> 4;

  for (int t = 0; t < T_; ++t) {
    const __bf16* hprev = (t == 0) ? H0 : (H + (size_t)(t - 1) * B_ * NH_);

    // ---- k = h @ W_key (+b_key): 4x2 tiles of 16x16, K split 4 ways ----
    {
      int p = wave >> 2, s = wave & 3;
      int mi = p & 3, ni = p >> 2;
      const __bf16* Ab = hprev + (size_t)(mi * 16 + lm) * NH_ + lh * 8;
      const __bf16* Bb = WkeyT + (size_t)(ni * 16 + lm) * NH_ + lh * 8;
      v8f c = {};
      int k0 = s * (NH_ / 4);
#pragma unroll 4
      for (int k = k0; k < k0 + NH_ / 4; k += 32)
        c = wmma_bf16(ldfrag(Ab + k), ldfrag(Bb + k), c);
#pragma unroll
      for (int r = 0; r < 8; ++r) partial[p][s][r * 32 + lane] = c[r];
    }
    __syncthreads();
    if (wave < 8) {
      int p = wave, mi = p & 3, ni = p >> 2;
#pragma unroll
      for (int r = 0; r < 8; ++r) {
        float v = partial[p][0][r * 32 + lane] + partial[p][1][r * 32 + lane]
                + partial[p][2][r * 32 + lane] + partial[p][3][r * 32 + lane];
        int row = mi * 16 + r + (lh << 3);
        int col = ni * 16 + lm;
        kbuf[row][col] = (col < MW_) ? (v + bkey[col]) : 0.0f;
      }
    }
    __syncthreads();

    // ---- cosine sim -> softmax -> r (2 batch rows per wave, 4 slots per lane) ----
    {
      const float* bank = mem + (size_t)(t & 3) * MN_ * MW_;
      const float* bn = norms + (t & 3) * MN_;
      for (int bi = 0; bi < 2; ++bi) {
        int b = wave * 2 + bi;
        float kn = 0.f;
#pragma unroll
        for (int c = 0; c < MW_; ++c) { float kv = kbuf[b][c]; kn += kv * kv; }
        kn = sqrtf(kn);
        float wv[4]; float mx = -3.4e38f;
#pragma unroll
        for (int q = 0; q < 4; ++q) {
          int j = lane + q * 32;
          const float* bp = bank + j * MW_;
          float d = 0.f;
#pragma unroll
          for (int c = 0; c < MW_; ++c) d += kbuf[b][c] * bp[c];
          float s = d / (kn * bn[j] + 1e-8f);
          wv[q] = s; mx = fmaxf(mx, s);
        }
#pragma unroll
        for (int o = 16; o > 0; o >>= 1) mx = fmaxf(mx, __shfl_xor(mx, o, 32));
        float sum = 0.f;
#pragma unroll
        for (int q = 0; q < 4; ++q) { wv[q] = __expf(wv[q] - mx); sum += wv[q]; }
#pragma unroll
        for (int o = 16; o > 0; o >>= 1) sum += __shfl_xor(sum, o, 32);
        float inv = 1.0f / sum;
        float racc[MW_];
#pragma unroll
        for (int c = 0; c < MW_; ++c) racc[c] = 0.f;
#pragma unroll
        for (int q = 0; q < 4; ++q) {
          int j = lane + q * 32;
          const float* bp = bank + j * MW_;
          float wq = wv[q] * inv;
#pragma unroll
          for (int c = 0; c < MW_; ++c) racc[c] += wq * bp[c];
        }
#pragma unroll
        for (int c = 0; c < MW_; ++c)
#pragma unroll
          for (int o = 16; o > 0; o >>= 1) racc[c] += __shfl_xor(racc[c], o, 32);
        if (lane == 0) {
#pragma unroll
          for (int c = 0; c < MW_; ++c) rbuf[b][c] = (__bf16)racc[c];
#pragma unroll
          for (int c = MW_; c < MWP_; ++c) rbuf[b][c] = (__bf16)0.0f;
        }
      }
    }
    __syncthreads();

    // ---- h_new = relu(Z[t] + r @ W_read + b_read); store bf16 to H[t] ----
    {
      const float* Zt = Z + (size_t)t * B_ * NH_;
      __bf16* Ht = H + (size_t)t * B_ * NH_;
      for (int it = 0; it < 16; ++it) {
        int q = wave + (it << 5);         // 0..511 = 4 m-tiles x 128 n-tiles
        int mi = q & 3, ni = q >> 2;
        v16bf a   = ldfrag(&rbuf[mi * 16 + lm][lh * 8]);
        v16bf bfr = ldfrag(WreadT + (size_t)(ni * 16 + lm) * MWP_ + lh * 8);
        v8f c = {};
        c = wmma_bf16(a, bfr, c);
#pragma unroll
        for (int r = 0; r < 8; ++r) {
          int row = mi * 16 + r + (lh << 3);
          int col = ni * 16 + lm;
          float v = c[r] + Zt[(size_t)row * NH_ + col] + bread[col];
          v = fmaxf(v, 0.0f);
          Ht[(size_t)row * NH_ + col] = (__bf16)v;
        }
      }
    }
    __threadfence();
    __syncthreads();
  }
}

// ---------------- launch ----------------
extern "C" void kernel_launch(void* const* d_in, const int* in_sizes, int n_in,
                              void* d_out, int out_size, void* d_ws, size_t ws_size,
                              hipStream_t stream) {
  const float* x      = (const float*)d_in[0];
  const float* h0     = (const float*)d_in[1];
  const float* W_in   = (const float*)d_in[2];
  const float* b_in   = (const float*)d_in[3];
  const float* W_read = (const float*)d_in[4];
  const float* b_read = (const float*)d_in[5];
  const float* W_out  = (const float*)d_in[6];
  const float* b_out  = (const float*)d_in[7];
  const float* W_key  = (const float*)d_in[8];
  const float* b_key  = (const float*)d_in[9];
  const float* memory = (const float*)d_in[10];
  float* out = (float*)d_out;
  (void)in_sizes; (void)n_in; (void)out_size; (void)ws_size;

  char* w = (char*)d_ws;
  size_t off = 0;
  auto alloc = [&](size_t bytes) {
    void* p = w + off;
    off = (off + bytes + 255) & ~(size_t)255;
    return p;
  };
  __bf16* Xbf    = (__bf16*)alloc((size_t)T_ * B_ * NI_ * 2);   // 16 MB
  __bf16* WinT   = (__bf16*)alloc((size_t)NH_ * NI_ * 2);       //  8 MB
  __bf16* WoutT  = (__bf16*)alloc((size_t)NI_ * NH_ * 2);       //  8 MB
  __bf16* WkeyT  = (__bf16*)alloc((size_t)MWP_ * NH_ * 2);      // 128 KB
  __bf16* WreadT = (__bf16*)alloc((size_t)NH_ * MWP_ * 2);      // 128 KB
  __bf16* H0bf   = (__bf16*)alloc((size_t)B_ * NH_ * 2);        // 256 KB
  __bf16* H      = (__bf16*)alloc((size_t)T_ * B_ * NH_ * 2);   // 16 MB
  float*  Z      = (float*)alloc((size_t)T_ * B_ * NH_ * 4);    // 32 MB
  float*  norms  = (float*)alloc((size_t)MB_ * MN_ * 4);        //  2 KB

  dim3 tb(32, 8);
  // phase 0: converts / transposes
  k_cvt_x<<<T_ * B_, 256, 0, stream>>>(x, Xbf);
  k_transpose_cvt<<<dim3(NH_ / 32, NI_ / 32), tb, 0, stream>>>(W_in,  WinT,  NI_, NH_, NI_, NH_);
  k_transpose_cvt<<<dim3(NI_ / 32, NH_ / 32), tb, 0, stream>>>(W_out, WoutT, NH_, NI_, NH_, NI_);
  k_transpose_cvt<<<dim3(1,        NH_ / 32), tb, 0, stream>>>(W_key, WkeyT, NH_, MW_, NH_, MWP_);
  k_transpose_cvt<<<dim3(NH_ / 32, 1),        tb, 0, stream>>>(W_read, WreadT, MW_, NH_, MWP_, NH_);
  k_cvt<<<(B_ * NH_ + 255) / 256, 256, 0, stream>>>(h0, H0bf, B_ * NH_);
  k_bank_norms<<<2, 256, 0, stream>>>(memory, norms);

  // phase 1: Z = X @ W_in + b_in   (4096x2048x2048, bf16 WMMA, f32 acc)
  k_gemm<<<256, 256, 0, stream>>>(Xbf, WinT, b_in, Z, T_ * B_, NH_, NI_, 0);

  // phase 2: sequential scan (k, softmax read, h_new) on one WGP
  k_recurrence<<<1, 1024, 0, stream>>>(H0bf, H, WkeyT, WreadT, b_key, b_read, Z, memory, norms);

  // phase 3: out = clip(H @ W_out + b_out, 0, 1), remapped to [B,T,NI]
  k_gemm<<<256, 256, 0, stream>>>(H, WoutT, b_out, out, T_ * B_, NI_, NH_, 1);
}
